// BinaryDilGroupConv_39444979646688
// MI455X (gfx1250) — 8-bit matrix operand order repaired and checked by reading, compiled, not run on hardware
//
#include <hip/hip_runtime.h>
#include <stdint.h>


typedef __attribute__((ext_vector_type(8))) int v8i;

#define CIN   256
#define HWPX  1024
#define WDIM  32
#define PW    36

__global__ __launch_bounds__(256)
void binconv_bn_shuffle_res(const float* __restrict__ xin,
                            const float* __restrict__ weight,
                            const float* __restrict__ gamma,
                            const float* __restrict__ beta,
                            const float* __restrict__ rmean,
                            const float* __restrict__ rvar,
                            float* __restrict__ out)
{
    __shared__ uint32_t smem[PW * PW * 4];

    const int tid    = threadIdx.x;
    const int lane   = tid & 31;
    const int wave   = tid >> 5;
    const int laneHi = (lane >> 4) & 1;
    const int blk    = blockIdx.x;
    const int gg     = blk & 15;
    const int n      = blk >> 4;
    const int cBase  = gg << 4;

    for (int i = tid; i < PW * PW * 4; i += 256) smem[i] = 0u;
    __syncthreads();

    for (int q = 0; q < 4; ++q) {
        float inv4[4], bias4[4];
        #pragma unroll
        for (int j = 0; j < 4; ++j) {
            const int c = cBase + q * 4 + j;
            const float iv = gamma[c] * rsqrtf(rvar[c] + 1e-5f);
            inv4[j]  = iv;
            bias4[j] = beta[c] - rmean[c] * iv;
        }
        #pragma unroll
        for (int it = 0; it < 4; ++it) {
            const int p = tid + it * 256;
            uint32_t packed = 0u;
            #pragma unroll
            for (int j = 0; j < 4; ++j) {
                const int c = cBase + q * 4 + j;
                const float h = xin[(n * CIN + c) * HWPX + p] * inv4[j] + bias4[j];
                const uint32_t s = (h >= 0.0f) ? 0x01u : 0xFFu;
                packed |= s << (8 * j);
            }
            const int py = (p >> 5) + 2;
            const int px = (p & 31) + 2;
            smem[(py * PW + px) * 4 + q] = packed;
        }
    }
    __syncthreads();

    const int co = lane & 15;
    v8i bfrag[3];
    #pragma unroll
    for (int w = 0; w < 3; ++w) {
        v8i bf;
        #pragma unroll
        for (int j = 0; j < 8; ++j) {
            const int quarter = j >> 1;
            uint32_t val = 0u;
            if (quarter < 3) {
                const int tap = 3 * w + quarter;
                #pragma unroll
                for (int b = 0; b < 4; ++b) {
                    const int ci = ((2 * laneHi + (j & 1)) << 2) + b;
                    if ((ci >> 2) == (co >> 2)) {
                        const float wv =
                            weight[((cBase + co) * 4 + (ci & 3)) * 9 + tap];
                        const uint32_t s = (wv >= 0.0f) ? 0x01u : 0xFFu;
                        val |= s << (8 * b);
                    }
                }
            }
            bf[j] = (int)val;
        }
        bfrag[w] = bf;
    }

    const int m = lane & 15;
    for (int tt = 0; tt < 8; ++tt) {
        const int t  = wave * 8 + tt;
        const int y  = t >> 1;
        const int x0 = (t & 1) << 4;

        v8i acc = {0, 0, 0, 0, 0, 0, 0, 0};
        #pragma unroll
        for (int w = 0; w < 3; ++w) {
            v8i af;
            #pragma unroll
            for (int j = 0; j < 6; ++j) {
                const int tap = 3 * w + (j >> 1);
                const int dy  = 2 * ((tap / 3) - 1);
                const int dx  = 2 * ((tap % 3) - 1);
                const int py  = y + dy + 2;
                const int px  = x0 + m + dx + 2;
                const int cq  = 2 * laneHi + (j & 1);
                af[j] = (int)smem[(py * PW + px) * 4 + cq];
            }
            af[6] = 0;
            af[7] = 0;
            acc = __builtin_amdgcn_wmma_i32_16x16x64_iu8(
                       true, af,  true, bfrag[w],
                      acc,  false,  false);
        }

        const int cs   = (co & 3) * 64 + (gg << 2) + (co >> 2);
        const int base = (n * CIN + cs) * HWPX + y * WDIM + x0 + 8 * laneHi;
        const float4* rin = (const float4*)(xin + base);
        float4*       po  = (float4*)(out + base);
        const float4 r0 = rin[0], r1 = rin[1];
        float4 o0, o1;
        o0.x = (float)acc[0] + r0.x;  o0.y = (float)acc[1] + r0.y;
        o0.z = (float)acc[2] + r0.z;  o0.w = (float)acc[3] + r0.w;
        o1.x = (float)acc[4] + r1.x;  o1.y = (float)acc[5] + r1.y;
        o1.z = (float)acc[6] + r1.z;  o1.w = (float)acc[7] + r1.w;
        po[0] = o0;
        po[1] = o1;
    }
}

extern "C" void kernel_launch(void* const* d_in, const int* in_sizes, int n_in,
                              void* d_out, int out_size, void* d_ws, size_t ws_size,
                              hipStream_t stream) {
    (void)n_in; (void)out_size; (void)d_ws; (void)ws_size;
    const float* x     = (const float*)d_in[0];
    const float* w     = (const float*)d_in[1];
    const float* gamma = (const float*)d_in[2];
    const float* beta  = (const float*)d_in[3];
    const float* rmean = (const float*)d_in[4];
    const float* rvar  = (const float*)d_in[5];
    float* out = (float*)d_out;

    const int N = in_sizes[0] / (CIN * HWPX);
    dim3 grid(N * 16);
    dim3 block(256);
    binconv_bn_shuffle_res<<<grid, block, 0, stream>>>(x, w, gamma, beta,
                                                       rmean, rvar, out);
}
